// GMMLoss_pca_82927228551272
// MI455X (gfx1250) — matrix operand order repaired and checked by reading, not compiled, not run on hardware
//
#include <hip/hip_runtime.h>
#include <hip/hip_bf16.h>

// ---------------------------------------------------------------------------
// Types for CDNA5 WMMA
// ---------------------------------------------------------------------------
typedef __attribute__((ext_vector_type(16))) __bf16 v16bf;
typedef __attribute__((ext_vector_type(8)))  float  v8f;

struct bfx16 { __bf16 v[16]; };

__device__ inline unsigned short bf16bits(float f) {
  unsigned u = __builtin_bit_cast(unsigned, f);
  unsigned r = u + 0x7FFFu + ((u >> 16) & 1u);   // round-to-nearest-even
  return (unsigned short)(r >> 16);
}
__device__ inline __bf16 f2bf(float f) {
  unsigned short h = bf16bits(f);
  return __builtin_bit_cast(__bf16, h);
}
// truncation-based hi/lo split: hi = f with low mantissa zeroed (exact residual)
__device__ inline void split_trunc(float f, __bf16& h, __bf16& l) {
  unsigned u  = __builtin_bit_cast(unsigned, f);
  unsigned hu = u & 0xFFFF0000u;
  float    hf = __builtin_bit_cast(float, hu);
  float    lf = f - hf;                          // exact
  unsigned lu = __builtin_bit_cast(unsigned, lf);
  h = __builtin_bit_cast(__bf16, (unsigned short)(hu >> 16));
  l = __builtin_bit_cast(__bf16, (unsigned short)(lu >> 16));
}

// Problem constants
#define BATCH 1024
#define FEAT  8192
#define KD    256     // padded subspace width (multiple of 64 -> no GEMM guards)
#define DIMR  200     // true PCA dim used in KL formulas
#define NC    8

// GEMM tile config
#define TM  64
#define TN  64
#define TK  32
#define TKP 40        // bf16 LDS row stride (elements): 80B, 16B-aligned
#define TKF 36        // fp32 LDS row stride (floats): 144B, 16B-aligned

#define MODE_NN      1   // C = A * B     (row-major)
#define MODE_TN      2   // C = A^T * B   (A is KxM row-major)
#define MODE_TN_MASK 3   // C_cls = A^T diag(mask_cls) B, batched over blockIdx.z

// ---------------------------------------------------------------------------
// CDNA5 async global->LDS copy (ASYNCcnt), 16 bytes per lane
// ---------------------------------------------------------------------------
__device__ inline void async_copy16(const float* gptr, void* lptr) {
  unsigned lds = (unsigned)(unsigned long long)lptr;
  asm volatile("global_load_async_to_lds_b128 %0, %1, off"
               :: "v"(lds), "v"((unsigned long long)gptr)
               : "memory");
}

// pack 8 floats -> 8 bf16 (one 16B LDS store)
__device__ inline void cvt_store8(__bf16* dst, const float* v) {
  unsigned q0 = (unsigned)bf16bits(v[0]) | ((unsigned)bf16bits(v[1]) << 16);
  unsigned q1 = (unsigned)bf16bits(v[2]) | ((unsigned)bf16bits(v[3]) << 16);
  unsigned q2 = (unsigned)bf16bits(v[4]) | ((unsigned)bf16bits(v[5]) << 16);
  unsigned q3 = (unsigned)bf16bits(v[6]) | ((unsigned)bf16bits(v[7]) << 16);
  *reinterpret_cast<uint4*>(dst) = make_uint4(q0, q1, q2, q3);
}

// A fragment (bf16 LDS): lane elements rows[8*hi .. +8) and rows[16+8*hi .. +8)
__device__ inline v16bf frag_a(const __bf16* rowp, int hi) {
  bfx16 t;
  __builtin_memcpy(&t.v[0], rowp + 8 * hi, 16);
  __builtin_memcpy(&t.v[8], rowp + 16 + 8 * hi, 16);
  return __builtin_bit_cast(v16bf, t);
}
// B fragment (bf16 LDS): lane elements rows[8*hi .. +8) and rows[16+8*hi .. +8)
__device__ inline v16bf frag_b(const __bf16* rowp, int hi) {
  bfx16 t;
  __builtin_memcpy(&t.v[0], rowp + 8 * hi, 16);
  __builtin_memcpy(&t.v[8], rowp + 16 + 8 * hi, 16);
  return __builtin_bit_cast(v16bf, t);
}

// fp32-LDS fragment builders with truncation hi/lo split (async Gram GEMM)
__device__ inline void frag_a_split(const float* rowp, int hi, v16bf& ah, v16bf& al) {
  float t[16];
  __builtin_memcpy(&t[0], rowp + 8 * hi, 32);
  __builtin_memcpy(&t[8], rowp + 16 + 8 * hi, 32);
  bfx16 H, L;
#pragma unroll
  for (int e = 0; e < 16; ++e) split_trunc(t[e], H.v[e], L.v[e]);
  ah = __builtin_bit_cast(v16bf, H);
  al = __builtin_bit_cast(v16bf, L);
}
__device__ inline void frag_b_split(const float* rowp, int hi, v16bf& bh, v16bf& bl) {
  float t[16];
  __builtin_memcpy(&t[0], rowp + 8 * hi, 32);
  __builtin_memcpy(&t[8], rowp + 16 + 8 * hi, 32);
  bfx16 H, L;
#pragma unroll
  for (int e = 0; e < 16; ++e) split_trunc(t[e], H.v[e], L.v[e]);
  bh = __builtin_bit_cast(v16bf, H);
  bl = __builtin_bit_cast(v16bf, L);
}

// ---------------------------------------------------------------------------
// Gram GEMM: C = A * B^T (A,B row-major xK), bf16 hi/lo split (3 WMMA/frag),
// async-to-LDS double-buffered fp32 staging. All dims multiples of tiles.
// ---------------------------------------------------------------------------
__global__ __launch_bounds__(256)
void gemm_gram_async(const float* __restrict__ A, const float* __restrict__ B,
                     float* __restrict__ C, int M, int N, int K,
                     int lda, int ldb, int ldc) {
  __shared__ float As32[2][TM][TKF];
  __shared__ float Bs32[2][TN][TKF];

  const int tid  = threadIdx.x;
  const int row0 = blockIdx.y * TM;
  const int col0 = blockIdx.x * TN;

  const int waveId = tid >> 5;
  const int lane   = tid & 31;
  const int wr  = waveId & 3;
  const int wc  = waveId >> 2;
  const int m16 = lane & 15;
  const int hi  = lane >> 4;

  v8f acc0 = {};
  v8f acc1 = {};

  // issue first tile (4 async b128 copies per thread: 2 for A, 2 for B)
#pragma unroll
  for (int c = 0; c < 2; ++c) {
    int f = tid + 256 * c;
    int m = f >> 3, kq = (f & 7) * 4;
    async_copy16(A + (size_t)(row0 + m) * lda + kq, &As32[0][m][kq]);
    async_copy16(B + (size_t)(col0 + m) * ldb + kq, &Bs32[0][m][kq]);
  }

  for (int k0 = 0; k0 < K; k0 += TK) {
    const int cur = (k0 / TK) & 1;
    const bool more = (k0 + TK) < K;
    if (more) {   // prefetch next tile into the other buffer via async DMA
#pragma unroll
      for (int c = 0; c < 2; ++c) {
        int f = tid + 256 * c;
        int m = f >> 3, kq = (f & 7) * 4;
        async_copy16(A + (size_t)(row0 + m) * lda + (k0 + TK) + kq, &As32[cur ^ 1][m][kq]);
        async_copy16(B + (size_t)(col0 + m) * ldb + (k0 + TK) + kq, &Bs32[cur ^ 1][m][kq]);
      }
      asm volatile("s_wait_asynccnt 4" ::: "memory");  // current tile landed
    } else {
      asm volatile("s_wait_asynccnt 0" ::: "memory");
    }
    __syncthreads();

    v16bf ah, al, b0h, b0l, b1h, b1l;
    frag_a_split(&As32[cur][wr * 16 + m16][0], hi, ah, al);
    frag_b_split(&Bs32[cur][wc * 32 + m16][0], hi, b0h, b0l);
    frag_b_split(&Bs32[cur][wc * 32 + 16 + m16][0], hi, b1h, b1l);
    acc0 = __builtin_amdgcn_wmma_f32_16x16x32_bf16(false, ah, false, b0h, (short)0, acc0, false, false);
    acc0 = __builtin_amdgcn_wmma_f32_16x16x32_bf16(false, ah, false, b0l, (short)0, acc0, false, false);
    acc0 = __builtin_amdgcn_wmma_f32_16x16x32_bf16(false, al, false, b0h, (short)0, acc0, false, false);
    acc1 = __builtin_amdgcn_wmma_f32_16x16x32_bf16(false, ah, false, b1h, (short)0, acc1, false, false);
    acc1 = __builtin_amdgcn_wmma_f32_16x16x32_bf16(false, ah, false, b1l, (short)0, acc1, false, false);
    acc1 = __builtin_amdgcn_wmma_f32_16x16x32_bf16(false, al, false, b1h, (short)0, acc1, false, false);
    __syncthreads();
  }

#pragma unroll
  for (int r = 0; r < 8; ++r) {
    int row = row0 + wr * 16 + hi * 8 + r;
    int col = col0 + wc * 32 + m16;
    C[(size_t)row * ldc + col]      = acc0[r];
    C[(size_t)row * ldc + col + 16] = acc1[r];
  }
}

// ---------------------------------------------------------------------------
// Small WMMA GEMMs (NN / TN / TN-masked), bf16 convert-on-store staging.
// ---------------------------------------------------------------------------
template <int MODE>
__global__ __launch_bounds__(256)
void gemm_wmma(const float* __restrict__ A, const float* __restrict__ B,
               float* __restrict__ C, const int* __restrict__ labels,
               int M, int N, int K, int lda, int ldb, int ldc, int batchStrideC) {
  __shared__ __bf16 As[TM][TKP];
  __shared__ __bf16 Bs[TN][TKP];

  const int tid  = threadIdx.x;
  const int row0 = blockIdx.y * TM;
  const int col0 = blockIdx.x * TN;
  const int cls  = blockIdx.z;
  if (MODE == MODE_TN_MASK) C += (size_t)cls * batchStrideC;

  const int waveId = tid >> 5;
  const int lane   = tid & 31;
  const int wr  = waveId & 3;
  const int wc  = waveId >> 2;
  const int m16 = lane & 15;
  const int hi  = lane >> 4;

  v8f acc0 = {};
  v8f acc1 = {};

  for (int k0 = 0; k0 < K; k0 += TK) {
    // ---------- A tile ----------
    if (MODE == MODE_NN) {
      int m  = tid >> 2;
      int kb = (tid & 3) * 8;
      const float* p = A + (size_t)(row0 + m) * lda + (k0 + kb);
      float4 f0 = *reinterpret_cast<const float4*>(p);
      float4 f1 = *reinterpret_cast<const float4*>(p + 4);
      __builtin_prefetch(p + TK, 0, 3);
      float v[8] = {f0.x, f0.y, f0.z, f0.w, f1.x, f1.y, f1.z, f1.w};
      cvt_store8(&As[m][kb], v);
    } else {  // TN / TN_MASK: A is KxM row-major, coalesced over m
      int k  = tid >> 3;
      int mb = (tid & 7) * 8;
      const float* p = A + (size_t)(k0 + k) * lda + (row0 + mb);
      float4 f0 = *reinterpret_cast<const float4*>(p);
      float4 f1 = *reinterpret_cast<const float4*>(p + 4);
      float msk = 1.f;
      if (MODE == MODE_TN_MASK) msk = (labels[k0 + k] == cls) ? 1.f : 0.f;
      float v[8] = {f0.x, f0.y, f0.z, f0.w, f1.x, f1.y, f1.z, f1.w};
#pragma unroll
      for (int u = 0; u < 8; ++u) As[mb + u][k] = f2bf(v[u] * msk);
    }
    // ---------- B tile (row-major KxN, coalesced over n) ----------
    {
      int k  = tid >> 3;
      int nb = (tid & 7) * 8;
      const float* p = B + (size_t)(k0 + k) * ldb + (col0 + nb);
      float4 f0 = *reinterpret_cast<const float4*>(p);
      float4 f1 = *reinterpret_cast<const float4*>(p + 4);
      float v[8] = {f0.x, f0.y, f0.z, f0.w, f1.x, f1.y, f1.z, f1.w};
#pragma unroll
      for (int u = 0; u < 8; ++u) Bs[nb + u][k] = f2bf(v[u]);
    }
    __syncthreads();

    v16bf a  = frag_a(&As[wr * 16 + m16][0], hi);
    v16bf b0 = frag_b(&Bs[wc * 32 + m16][0], hi);
    v16bf b1 = frag_b(&Bs[wc * 32 + 16 + m16][0], hi);
    acc0 = __builtin_amdgcn_wmma_f32_16x16x32_bf16(false, a, false, b0, (short)0, acc0, false, false);
    acc1 = __builtin_amdgcn_wmma_f32_16x16x32_bf16(false, a, false, b1, (short)0, acc1, false, false);
    __syncthreads();
  }

#pragma unroll
  for (int r = 0; r < 8; ++r) {
    int row = row0 + wr * 16 + hi * 8 + r;
    int col = col0 + wc * 32 + m16;
    C[(size_t)row * ldc + col]      = acc0[r];
    C[(size_t)row * ldc + col + 16] = acc1[r];
  }
}

// ---------------------------------------------------------------------------
// Mean statistics
// ---------------------------------------------------------------------------
__global__ void colmean_kernel(const float* __restrict__ X, float* __restrict__ mean) {
  int j = blockIdx.x * blockDim.x + threadIdx.x;
  if (j >= FEAT) return;
  float s = 0.f;
  for (int i = 0; i < BATCH; ++i) s += X[(size_t)i * FEAT + j];
  mean[j] = s * (1.0f / BATCH);
}

__global__ void rowdot_kernel(const float* __restrict__ X, const float* __restrict__ mean,
                              float* __restrict__ r) {
  __shared__ float red[256];
  int i = blockIdx.x;
  float s = 0.f;
  for (int k = threadIdx.x; k < FEAT; k += 256) s += X[(size_t)i * FEAT + k] * mean[k];
  red[threadIdx.x] = s;
  __syncthreads();
  for (int off = 128; off > 0; off >>= 1) {
    if ((int)threadIdx.x < off) red[threadIdx.x] += red[threadIdx.x + off];
    __syncthreads();
  }
  if (threadIdx.x == 0) r[i] = red[0];
}

__global__ void msq_kernel(const float* __restrict__ mean, float* __restrict__ msq) {
  __shared__ float red[256];
  float s = 0.f;
  for (int k = threadIdx.x; k < FEAT; k += 256) s += mean[k] * mean[k];
  red[threadIdx.x] = s;
  __syncthreads();
  for (int off = 128; off > 0; off >>= 1) {
    if ((int)threadIdx.x < off) red[threadIdx.x] += red[threadIdx.x + off];
    __syncthreads();
  }
  if (threadIdx.x == 0) msq[0] = red[0];
}

// G = XX^T - r 1^T - 1 r^T + (m.m) J
__global__ void gram_correct_kernel(float* __restrict__ G, const float* __restrict__ r,
                                    const float* __restrict__ msq) {
  int idx = blockIdx.x * blockDim.x + threadIdx.x;
  if (idx >= BATCH * BATCH) return;
  int i = idx / BATCH, j = idx % BATCH;
  G[idx] = G[idx] - r[i] - r[j] + msq[0];
}

// ---------------------------------------------------------------------------
// Deterministic pseudo-random subspace init
// ---------------------------------------------------------------------------
__device__ inline unsigned wang(unsigned s) {
  s = (s ^ 61u) ^ (s >> 16); s *= 9u; s ^= s >> 4; s *= 0x27d4eb2du; s ^= s >> 15;
  return s;
}
__global__ void init_q_kernel(float* __restrict__ Q) {
  int idx = blockIdx.x * blockDim.x + threadIdx.x;
  if (idx >= BATCH * KD) return;
  unsigned h = wang((unsigned)idx * 2654435761u + 12345u);
  Q[idx] = ((float)(h & 0xFFFFFFu) / 16777216.0f) - 0.5f;
}

// ---------------------------------------------------------------------------
// In-place lower Cholesky, one workgroup per matrix
// ---------------------------------------------------------------------------
__global__ void cholesky_kernel(float* __restrict__ Abase, int batchStride, int N) {
  float* A = Abase + (size_t)blockIdx.x * batchStride;
  int tid = threadIdx.x;
  for (int k = 0; k < N; ++k) {
    __syncthreads();
    if (tid == 0) A[k * N + k] = sqrtf(fmaxf(A[k * N + k], 1e-20f));
    __syncthreads();
    float dk = A[k * N + k];
    for (int i = k + 1 + tid; i < N; i += blockDim.x) A[i * N + k] /= dk;
    __syncthreads();
    for (int i = k + 1 + tid; i < N; i += blockDim.x) {
      float lik = A[i * N + k];
      for (int j = k + 1; j <= i; ++j) A[i * N + j] -= lik * A[j * N + k];
    }
  }
}

// ---------------------------------------------------------------------------
// Cholesky-QR row solve: Q[i,:] = L^{-1} Y[i,:] (forward substitution per row)
// ---------------------------------------------------------------------------
__global__ void rowsolve_kernel(const float* __restrict__ L, const float* __restrict__ Y,
                                float* __restrict__ Q, int rows, int n) {
  extern __shared__ float wsh[];  // blockDim.x * (n+1)
  int tid = threadIdx.x;
  int row = blockIdx.x * blockDim.x + tid;
  float* w = wsh + (size_t)tid * (n + 1);
  if (row < rows) {
    for (int j = 0; j < n; ++j) {
      float s = Y[(size_t)row * n + j];
      for (int t = 0; t < j; ++t) s -= L[j * n + t] * w[t];
      w[j] = s / L[j * n + j];
    }
    for (int j = 0; j < n; ++j) Q[(size_t)row * n + j] = w[j];
  }
}

// ---------------------------------------------------------------------------
// lam[j] = sum_i Q[i][j] * Y[i][j]
// ---------------------------------------------------------------------------
__global__ void lambda_kernel(const float* __restrict__ Q, const float* __restrict__ Y,
                              float* __restrict__ lam) {
  __shared__ float red[256];
  int j = blockIdx.x;
  float s = 0.f;
  for (int i = threadIdx.x; i < BATCH; i += 256)
    s += Q[(size_t)i * KD + j] * Y[(size_t)i * KD + j];
  red[threadIdx.x] = s;
  __syncthreads();
  for (int off = 128; off > 0; off >>= 1) {
    if ((int)threadIdx.x < off) red[threadIdx.x] += red[threadIdx.x + off];
    __syncthreads();
  }
  if (threadIdx.x == 0) lam[j] = red[0];
}

__global__ void sort_kernel(const float* __restrict__ lam, int* __restrict__ perm) {
  if (threadIdx.x != 0) return;
  bool used[KD];
  for (int i = 0; i < KD; ++i) used[i] = false;
  for (int s = 0; s < KD; ++s) {
    int best = 0; float bv = -3.4e38f;
    for (int t = 0; t < KD; ++t)
      if (!used[t] && lam[t] > bv) { bv = lam[t]; best = t; }
    used[best] = true;
    perm[s] = best;
  }
}

__global__ void scale_x_kernel(const float* __restrict__ Q, const float* __restrict__ lam,
                               const int* __restrict__ perm, float* __restrict__ x) {
  int idx = blockIdx.x * blockDim.x + threadIdx.x;
  if (idx >= BATCH * KD) return;
  int i = idx / KD, j = idx % KD;
  float v = 0.f;
  if (j < DIMR) {
    int pj = perm[j];
    v = Q[(size_t)i * KD + pj] * sqrtf(fmaxf(lam[pj], 0.f));
  }
  x[idx] = v;
}

// ---------------------------------------------------------------------------
// Per-class counts and means
// ---------------------------------------------------------------------------
__global__ void class_stats_kernel(const float* __restrict__ x, const int* __restrict__ labels,
                                   float* __restrict__ means8, float* __restrict__ counts) {
  __shared__ float cnt_sh;
  int l = blockIdx.x, tid = threadIdx.x;
  if (tid == 0) {
    int c = 0;
    for (int i = 0; i < BATCH; ++i) c += (labels[i] == l);
    cnt_sh = (float)c;
    counts[l] = (float)c;
  }
  __syncthreads();
  float csafe = fmaxf(cnt_sh, 1.f);
  for (int d = tid; d < KD; d += blockDim.x) {
    float s = 0.f;
    for (int i = 0; i < BATCH; ++i)
      if (labels[i] == l) s += x[(size_t)i * KD + d];
    means8[l * KD + d] = s / csafe;
  }
}

__global__ void center_kernel(const float* __restrict__ x, const int* __restrict__ labels,
                              const float* __restrict__ means8, float* __restrict__ cen) {
  int idx = blockIdx.x * blockDim.x + threadIdx.x;
  if (idx >= BATCH * KD) return;
  int i = idx / KD, d = idx % KD;
  cen[idx] = x[idx] - means8[labels[i] * KD + d];
}

__global__ void finalize_sigma_kernel(float* __restrict__ sig, const float* __restrict__ counts) {
  int idx = blockIdx.x * blockDim.x + threadIdx.x;
  if (idx >= NC * KD * KD) return;
  int l = idx / (KD * KD);
  int r = (idx / KD) % KD;
  int c = idx % KD;
  float cs = fmaxf(counts[l], 1.f);
  sig[idx] = sig[idx] / cs + ((r == c) ? 1.f : 0.f);
}

// ---------------------------------------------------------------------------
// KL(i,j)*count_i*count_j per pair.  tr = ||L_j^{-1}L_i||_F^2, maha = ||L_j^{-1}d||^2
// ---------------------------------------------------------------------------
__global__ void kl_kernel(const float* __restrict__ chol, const float* __restrict__ means8,
                          const float* __restrict__ counts, float* __restrict__ klw) {
  extern __shared__ float sh[];
  const int N = KD, D = DIMR;
  int p = blockIdx.x, i = p / NC, j = p % NC;
  int tid = threadIdx.x;
  float* w   = sh + (size_t)tid * (D + 1);
  float* red = sh + (size_t)blockDim.x * (D + 1);
  const float* Li = chol + (size_t)i * N * N;
  const float* Lj = chol + (size_t)j * N * N;

  float tr = 0.f;
  if (i != j) {
    for (int c = tid; c < D; c += blockDim.x) {
      for (int t = 0; t < D; ++t) {
        float s = (t >= c) ? Li[t * N + c] : 0.f;   // lower triangle only
        for (int u = 0; u < t; ++u) s -= Lj[t * N + u] * w[u];
        w[t] = s / Lj[t * N + t];
        tr += w[t] * w[t];
      }
    }
  }
  red[tid] = tr;
  __syncthreads();
  for (int off = blockDim.x / 2; off > 0; off >>= 1) {
    if (tid < off) red[tid] += red[tid + off];
    __syncthreads();
  }
  if (tid == 0) {
    float out = 0.f;
    if (i != j) {
      float trace = red[0];
      float maha = 0.f;
      for (int t = 0; t < D; ++t) {
        float s = means8[j * N + t] - means8[i * N + t];
        for (int u = 0; u < t; ++u) s -= Lj[t * N + u] * w[u];
        w[t] = s / Lj[t * N + t];
        maha += w[t] * w[t];
      }
      float ldi = 0.f, ldj = 0.f;
      for (int t = 0; t < D; ++t) {
        ldi += logf(Li[t * N + t]);
        ldj += logf(Lj[t * N + t]);
      }
      float kl = 0.5f * (2.f * ldj - 2.f * ldi - (float)D + trace + maha);
      out = kl * counts[i] * counts[j];
    }
    klw[p] = out;
  }
}

__global__ void final_reduce_kernel(const float* __restrict__ klw, float* __restrict__ out) {
  __shared__ float red[64];
  int tid = threadIdx.x;
  red[tid] = klw[tid];
  __syncthreads();
  for (int off = 32; off > 0; off >>= 1) {
    if (tid < off) red[tid] += red[tid + off];
    __syncthreads();
  }
  if (tid == 0) out[0] = red[0] / ((float)BATCH * (float)BATCH);
}

// ---------------------------------------------------------------------------
// Host orchestration
// ---------------------------------------------------------------------------
extern "C" void kernel_launch(void* const* d_in, const int* in_sizes, int n_in,
                              void* d_out, int out_size, void* d_ws, size_t ws_size,
                              hipStream_t stream) {
  (void)in_sizes; (void)n_in; (void)out_size; (void)ws_size;
  const float* mu     = (const float*)d_in[0];
  const int*   labels = (const int*)d_in[1];
  float*       out    = (float*)d_out;

  float* ws     = (float*)d_ws;
  float* mean   = ws;  ws += FEAT;
  float* rvec   = ws;  ws += BATCH;
  float* msq    = ws;  ws += 1;
  float* G      = ws;  ws += BATCH * BATCH;
  float* Q      = ws;  ws += BATCH * KD;
  float* Y      = ws;  ws += BATCH * KD;
  float* Mm     = ws;  ws += KD * KD;
  float* x      = ws;  ws += BATCH * KD;
  float* cen    = ws;  ws += BATCH * KD;
  float* means8 = ws;  ws += NC * KD;
  float* counts = ws;  ws += NC;
  float* sig    = ws;  ws += NC * KD * KD;
  float* lam    = ws;  ws += KD;
  float* klw    = ws;  ws += 64;
  int*   perm   = (int*)ws;

  const int NT = KD / TN;        // 4
  const int MT = BATCH / TM;     // 16

  // 1) mean stats + async-staged Gram GEMM, then exact rank-1 centering fixup
  colmean_kernel<<<dim3(FEAT / 256), dim3(256), 0, stream>>>(mu, mean);
  rowdot_kernel<<<dim3(BATCH), dim3(256), 0, stream>>>(mu, mean, rvec);
  msq_kernel<<<dim3(1), dim3(256), 0, stream>>>(mean, msq);
  gemm_gram_async<<<dim3(MT, MT, 1), dim3(256), 0, stream>>>(
      mu, mu, G, BATCH, BATCH, FEAT, FEAT, FEAT, BATCH);
  gram_correct_kernel<<<dim3(BATCH * BATCH / 256), dim3(256), 0, stream>>>(G, rvec, msq);

  // 2) subspace iteration with Cholesky-QR for top-KD eigenvectors of G
  init_q_kernel<<<dim3(BATCH * KD / 256), dim3(256), 0, stream>>>(Q);
  for (int it = 0; it < 10; ++it) {
    gemm_wmma<MODE_NN><<<dim3(NT, MT, 1), dim3(256), 0, stream>>>(
        G, Q, Y, nullptr, BATCH, KD, BATCH, BATCH, KD, KD, 0);
    gemm_wmma<MODE_TN><<<dim3(NT, NT, 1), dim3(256), 0, stream>>>(
        Y, Y, Mm, nullptr, KD, KD, BATCH, KD, KD, KD, 0);
    cholesky_kernel<<<dim3(1), dim3(256), 0, stream>>>(Mm, 0, KD);
    rowsolve_kernel<<<dim3(BATCH / 128), dim3(128), 128 * (KD + 1) * sizeof(float), stream>>>(
        Mm, Y, Q, BATCH, KD);
  }

  // 3) eigenvalues, sort desc, PCA scores x = V_k * sqrt(lambda_k)
  gemm_wmma<MODE_NN><<<dim3(NT, MT, 1), dim3(256), 0, stream>>>(
      G, Q, Y, nullptr, BATCH, KD, BATCH, BATCH, KD, KD, 0);
  lambda_kernel<<<dim3(KD), dim3(256), 0, stream>>>(Q, Y, lam);
  sort_kernel<<<dim3(1), dim3(32), 0, stream>>>(lam, perm);
  scale_x_kernel<<<dim3(BATCH * KD / 256), dim3(256), 0, stream>>>(Q, lam, perm, x);

  // 4) per-class stats, masked covariance GEMMs batched over blockIdx.z
  class_stats_kernel<<<dim3(NC), dim3(256), 0, stream>>>(x, labels, means8, counts);
  center_kernel<<<dim3(BATCH * KD / 256), dim3(256), 0, stream>>>(x, labels, means8, cen);
  gemm_wmma<MODE_TN_MASK><<<dim3(NT, NT, NC), dim3(256), 0, stream>>>(
      cen, cen, sig, labels, KD, KD, BATCH, KD, KD, KD, KD * KD);
  finalize_sigma_kernel<<<dim3(NC * KD * KD / 256), dim3(256), 0, stream>>>(sig, counts);

  // 5) batched Cholesky (in place), pairwise KL, final reduction
  cholesky_kernel<<<dim3(NC), dim3(256), 0, stream>>>(sig, KD * KD, KD);
  kl_kernel<<<dim3(NC * NC), dim3(128),
              (128 * (DIMR + 1) + 128) * sizeof(float), stream>>>(sig, means8, counts, klw);
  final_reduce_kernel<<<dim3(1), dim3(64), 0, stream>>>(klw, out);
}
